// MetaMoE_28089086116699
// MI455X (gfx1250) — matrix operand order repaired and checked by reading, not compiled, not run on hardware
//
#include <hip/hip_runtime.h>
#include <hip/hip_bf16.h>
#include <math.h>

// ---------------------------------------------------------------------------
// MetaMoE forward for MI455X (gfx1250): bf16 WMMA (v_wmma_f32_16x16x32_bf16),
// pre-transposed bf16 weights so all LDS traffic is b128-vectorized,
// TDM (tensor_load_to_lds) for the GEMM2 weight tile, wave32 layouts per
// CDNA5 ISA 7.12.2.
// ---------------------------------------------------------------------------

typedef __bf16 bf16;
typedef bf16         v8bf  __attribute__((ext_vector_type(8)));
typedef bf16         v16bf __attribute__((ext_vector_type(16)));
typedef float        v8f   __attribute__((ext_vector_type(8)));
typedef unsigned int v4u   __attribute__((ext_vector_type(4)));
typedef unsigned int v8u   __attribute__((ext_vector_type(8)));

#define NE   16
#define NIN  1024
#define NHID 2048
#define NB   4096
#define NG   256      // gate hidden
#define EPSF 1e-5f

// ---------------- WMMA helpers (CDNA5 16x16x32 bf16, wave32) ----------------

__device__ __forceinline__ v8f wmma_bf16(v16bf a, v16bf b, v8f c) {
  return __builtin_amdgcn_wmma_f32_16x16x32_bf16(false, a, false, b,
                                                 (short)0, c, false, false);
}

// A fragment: 16(M) x 32(K) bf16, row-major in LDS, row stride ldk elems.
__device__ __forceinline__ v16bf frag_a(const bf16* base, int lane, int ldk) {
  int row = lane & 15;
  int kb  = (lane >> 4) << 3;          // 0 or 8
  const bf16* p = base + row * ldk + kb;
  union { v16bf v; v8bf h[2]; } u;
  u.h[0] = *(const v8bf*)(p);
  u.h[1] = *(const v8bf*)(p + 16);
  return u.v;
}

// B fragment from LDS tile stored n-major: Bt[n][k], k-stride 32 elems.
__device__ __forceinline__ v16bf frag_bt(const bf16* base, int lane) {
  int col = lane & 15;
  int kb  = (lane >> 4) << 3;          // 0 or 8
  const bf16* p = base + col * 32 + kb;
  union { v16bf v; v8bf h[2]; } u;
  u.h[0] = *(const v8bf*)(p);
  u.h[1] = *(const v8bf*)(p + 16);
  return u.v;
}

// A-tile cooperative load with fused affine (v*g+b), 128x32 bf16, vectorized.
__device__ __forceinline__ void load_a_tile(bf16* lA, const bf16* xn, int row0,
                                            int kk, const float* gg0,
                                            const float* bb0, int tid) {
  int r  = tid >> 1;
  int c0 = (tid & 1) * 16;
  const bf16*  src = xn + (size_t)(row0 + r) * NIN + kk + c0;
  const float* gg  = gg0 + kk + c0;
  const float* bb  = bb0 + kk + c0;
  v8bf xa = ((const v8bf*)src)[0];
  v8bf xb = ((const v8bf*)src)[1];
  union { v8bf v; bf16 s[8]; } o0, o1;
#pragma unroll
  for (int i = 0; i < 8; ++i) o0.s[i] = (bf16)((float)xa[i] * gg[i] + bb[i]);
#pragma unroll
  for (int i = 0; i < 8; ++i) o1.s[i] = (bf16)((float)xb[i] * gg[8 + i] + bb[8 + i]);
  v8bf* d = (v8bf*)(lA + r * 32 + c0);
  d[0] = o0.v;
  d[1] = o1.v;
}

// ---------------- transpose + convert: in (K,N) f32 -> out (N,K) bf16 -------
__global__ __launch_bounds__(256) void tcvt_kernel(const float* __restrict__ in,
                                                   bf16* __restrict__ out,
                                                   int K, int N) {
  __shared__ float t[32][33];
  const float* src = in + (size_t)blockIdx.z * K * N;
  bf16*        dst = out + (size_t)blockIdx.z * K * N;
  int tx = threadIdx.x & 31;
  int ty = threadIdx.x >> 5;
  int k0 = blockIdx.y * 32;
  int n0 = blockIdx.x * 32;
#pragma unroll
  for (int i = 0; i < 32; i += 8)
    t[ty + i][tx] = src[(size_t)(k0 + ty + i) * N + n0 + tx];
  __syncthreads();
#pragma unroll
  for (int i = 0; i < 32; i += 8)
    dst[(size_t)(n0 + ty + i) * K + k0 + tx] = (bf16)t[tx][ty + i];
}

__global__ __launch_bounds__(256) void zero_f32(float* __restrict__ p, long n) {
  long i = (long)blockIdx.x * blockDim.x + threadIdx.x;
  long stride = (long)gridDim.x * blockDim.x;
  for (; i < n; i += stride) p[i] = 0.0f;
}

// LayerNorm: one block per row of x (4096 x 1024), output bf16 xn.
__global__ __launch_bounds__(256) void ln_kernel(const float* __restrict__ x,
                                                 bf16* __restrict__ xn) {
  __shared__ float s1[256];
  __shared__ float s2[256];
  int row = blockIdx.x;
  int tid = threadIdx.x;
  const float* xr = x + (size_t)row * NIN;
  float v[4], sum = 0.f, sq = 0.f;
#pragma unroll
  for (int i = 0; i < 4; ++i) {
    v[i] = xr[tid + i * 256];
    sum += v[i];
    sq  += v[i] * v[i];
  }
  s1[tid] = sum; s2[tid] = sq;
  __syncthreads();
  for (int s = 128; s > 0; s >>= 1) {
    if (tid < s) { s1[tid] += s1[tid + s]; s2[tid] += s2[tid + s]; }
    __syncthreads();
  }
  float mu  = s1[0] * (1.0f / NIN);
  float var = s2[0] * (1.0f / NIN) - mu * mu;
  float rs  = rsqrtf(var + EPSF);
#pragma unroll
  for (int i = 0; i < 4; ++i)
    xn[(size_t)row * NIN + tid + i * 256] = (bf16)((v[i] - mu) * rs);
}

// ------- gating GEMM1: g = relu((xn*g_ln_g+g_ln_b) @ g_w1 + g_b1) -----------
// w1t is pre-transposed: (N=256, K=1024) bf16. M tile 128/block, 8 waves.
__global__ __launch_bounds__(256) void gate_gemm1(const bf16* __restrict__ xn,
                                                  const float* __restrict__ lng,
                                                  const float* __restrict__ lnb,
                                                  const bf16* __restrict__ w1t,
                                                  const float* __restrict__ b1,
                                                  bf16* __restrict__ gout) {
  __shared__ __align__(16) bf16 lA[128 * 32];   // 8 KB
  __shared__ __align__(16) bf16 lBt[256 * 32];  // 16 KB
  int tid  = threadIdx.x;
  int wave = tid >> 5;
  int lane = tid & 31;
  int row0 = blockIdx.x * 128;

  v8f acc[16];
#pragma unroll
  for (int i = 0; i < 16; ++i) acc[i] = v8f{0.f,0.f,0.f,0.f,0.f,0.f,0.f,0.f};

  for (int kk = 0; kk < NIN; kk += 32) {
    load_a_tile(lA, xn, row0, kk, lng, lnb, tid);
    {   // B tile: w1t[n][kk..kk+32], n = tid  (fully vectorized)
      const bf16* src = w1t + (size_t)tid * NIN + kk;
      __builtin_prefetch(src + 32, 0, 1);
      const v8bf* s = (const v8bf*)src;
      v8bf* d = (v8bf*)(lBt + tid * 32);
      d[0] = s[0]; d[1] = s[1]; d[2] = s[2]; d[3] = s[3];
    }
    __syncthreads();
    v16bf a = frag_a(lA + wave * 16 * 32, lane, 32);
#pragma unroll
    for (int nt = 0; nt < 16; ++nt) {
      v16bf b = frag_bt(lBt + nt * 16 * 32, lane);
      acc[nt] = wmma_bf16(a, b, acc[nt]);
    }
    __syncthreads();
  }

  int hi8 = (lane >> 4) << 3;
  int cl  = lane & 15;
#pragma unroll
  for (int nt = 0; nt < 16; ++nt) {
    float bias = b1[nt * 16 + cl];
#pragma unroll
    for (int r = 0; r < 8; ++r) {
      int row = row0 + wave * 16 + r + hi8;
      gout[(size_t)row * NG + nt * 16 + cl] = (bf16)fmaxf(acc[nt][r] + bias, 0.f);
    }
  }
}

// gating layer 2: logits[b,e] = g[b,:] . g_w2[:,e] + g_b2[e]
__global__ __launch_bounds__(256) void gate_logits(const bf16* __restrict__ g,
                                                   const float* __restrict__ w2,
                                                   const float* __restrict__ b2,
                                                   float* __restrict__ logits) {
  int idx = blockIdx.x * 256 + threadIdx.x;
  if (idx >= NB * NE) return;
  int b = idx >> 4, e = idx & 15;
  const bf16* gr = g + (size_t)b * NG;
  float s = b2[e];
  for (int j = 0; j < NG; ++j) s += (float)gr[j] * w2[j * NE + e];
  logits[idx] = s;
}

__global__ __launch_bounds__(256) void gate_softmax(const float* __restrict__ logits,
                                                    float* __restrict__ gw) {
  int b = blockIdx.x * 256 + threadIdx.x;
  if (b >= NB) return;
  const float* lr = logits + b * NE;
  float mx = lr[0];
#pragma unroll
  for (int j = 1; j < NE; ++j) mx = fmaxf(mx, lr[j]);
  float ex[NE], s = 0.f;
#pragma unroll
  for (int j = 0; j < NE; ++j) { ex[j] = expf(lr[j] - mx); s += ex[j]; }
  float inv = 1.0f / s;
#pragma unroll
  for (int j = 0; j < NE; ++j) gw[b * NE + j] = ex[j] * inv;
}

// ---------------- fused expert kernel ----------------
// grid (B/128, E); w1t (E, 2048, 1024) bf16, w2t (E, 256, 2048) bf16.
__global__ __launch_bounds__(256) void expert_kernel(const bf16* __restrict__ xn,
                                                     const float* __restrict__ elng,
                                                     const float* __restrict__ elnb,
                                                     const bf16* __restrict__ w1tb,
                                                     const float* __restrict__ b1,
                                                     const bf16* __restrict__ w2tb,
                                                     const float* __restrict__ b2,
                                                     const float* __restrict__ w3,
                                                     const float* __restrict__ b3,
                                                     const float* __restrict__ gate_w,
                                                     float* __restrict__ out_acc) {
  // 48 KB LDS: [ lA(8K) | lB1(8K) ] union [ lB2(16K) ]  then  lH1(32K)
  __shared__ __align__(16) unsigned char smem[49152];
  bf16* lA  = (bf16*)smem;                 // 128 x 32
  bf16* lB1 = (bf16*)(smem + 8192);        // 128 x 32  (w1t chunk, n-major)
  bf16* lB2 = (bf16*)smem;                 // 256 x 32  (w2t chunk, n-major)
  bf16* lH1 = (bf16*)(smem + 16384);       // 128 x 128 (h1 chunk, bf16)

  int tid  = threadIdx.x;
  int wave = tid >> 5;
  int lane = tid & 31;
  int row0 = blockIdx.x * 128;
  int e    = blockIdx.y;

  const bf16*  w1te = w1tb + (size_t)e * NIN * NHID;   // (2048 n, 1024 k)
  const bf16*  w2te = w2tb + (size_t)e * NHID * NG;    // (256 n, 2048 k)
  const float* eg   = elng + (size_t)e * NIN;
  const float* eb   = elnb + (size_t)e * NIN;
  const float* b1e  = b1 + (size_t)e * NHID;
  const float* b2e  = b2 + (size_t)e * NG;
  const float* w3e  = w3 + (size_t)e * NG * 2;

  const v8f vz = v8f{0.f,0.f,0.f,0.f,0.f,0.f,0.f,0.f};
  v8f acc2[16];                            // h2: wave's 16 rows x 256 cols
#pragma unroll
  for (int i = 0; i < 16; ++i) acc2[i] = vz;

  int hi8 = (lane >> 4) << 3;
  int cl  = lane & 15;

  for (int hh = 0; hh < NHID; hh += 128) {
    v8f acc1[8];                           // h1 chunk: 16 rows x 128 cols
#pragma unroll
    for (int i = 0; i < 8; ++i) acc1[i] = vz;

    // ---- GEMM1: xe(128x1024) @ w1[:,hh:hh+128] ----
    for (int kk = 0; kk < NIN; kk += 32) {
      load_a_tile(lA, xn, row0, kk, eg, eb, tid);
      {   // w1t tile: rows hh..hh+128, k window 32 (vectorized, no transpose)
        int n  = tid >> 1;
        int k0 = (tid & 1) * 16;
        const bf16* src = w1te + (size_t)(hh + n) * NIN + kk + k0;
        __builtin_prefetch(src + 32, 0, 1);
        const v8bf* s = (const v8bf*)src;
        v8bf* d = (v8bf*)(lB1 + n * 32 + k0);
        d[0] = s[0]; d[1] = s[1];
      }
      __syncthreads();
      v16bf a = frag_a(lA + wave * 16 * 32, lane, 32);
#pragma unroll
      for (int nt = 0; nt < 8; ++nt) {
        v16bf b = frag_bt(lB1 + nt * 16 * 32, lane);
        acc1[nt] = wmma_bf16(a, b, acc1[nt]);
      }
      __syncthreads();
    }

    // ---- bias + relu + bf16 -> lH1 ----
#pragma unroll
    for (int nt = 0; nt < 8; ++nt) {
      float bias = b1e[hh + nt * 16 + cl];
#pragma unroll
      for (int r = 0; r < 8; ++r) {
        float v = fmaxf(acc1[nt][r] + bias, 0.f);
        lH1[(wave * 16 + r + hi8) * 128 + nt * 16 + cl] = (bf16)v;
      }
    }
    __syncthreads();

    // ---- GEMM2: h2 += h1c(128x128) @ w2[hh:hh+128, 0:256] ----
    for (int k2 = 0; k2 < 128; k2 += 32) {
      // w2t tile 256(n) x 32(k) via Tensor Data Mover (CDNA5 ISA ch.7/8):
      // D# group0: count=1, lds_addr, global_addr, type=2
      // D# group1: data_size=2B, tensor 2048x256, tile 32x256, stride0=2048
      if (wave == 0) {
        unsigned long long ga =
            (unsigned long long)(uintptr_t)(w2te + (size_t)(hh + k2));
        unsigned int lo = (unsigned int)(uintptr_t)lB2;  // LDS byte offset
        v4u g0;
        g0[0] = 1u;                                   // count=1 (valid D#)
        g0[1] = lo;                                   // lds_addr
        g0[2] = (unsigned int)ga;                     // global_addr[31:0]
        g0[3] = (unsigned int)(ga >> 32) | (2u << 30);// ga[56:32] | type=2
        v8u g1;
        g1[0] = 1u << 16;                             // data_size=1 -> 2 bytes
        g1[1] = (unsigned)(NHID & 0xFFFF) << 16;      // tensor_dim0[15:0]
        g1[2] = (unsigned)(NHID >> 16) | ((unsigned)(NG & 0xFFFF) << 16);
        g1[3] = (unsigned)(NG >> 16) | (32u << 16);   // | tile_dim0=32
        g1[4] = (unsigned)NG;                         // tile_dim1=256, tile_dim2=0
        g1[5] = (unsigned)NHID;                       // tensor_dim0_stride[31:0]
        g1[6] = 0u;
        g1[7] = 0u;
        asm volatile("tensor_load_to_lds %0, %1" :: "s"(g0), "s"(g1) : "memory");
        __builtin_amdgcn_s_wait_tensorcnt(0);
      }
      __syncthreads();
      v16bf a = frag_a(lH1 + (size_t)(wave * 16) * 128 + k2, lane, 128);
#pragma unroll
      for (int nt = 0; nt < 16; ++nt) {
        v16bf b = frag_bt(lB2 + nt * 16 * 32, lane);
        acc2[nt] = wmma_bf16(a, b, acc2[nt]);
      }
      __syncthreads();
    }
  }

  // ---- epilogue: eo = relu(h2+b2) @ w3 + b3, gated atomic accumulate ----
  float b2v[16], w3c0[16], w3c1[16];
#pragma unroll
  for (int nt = 0; nt < 16; ++nt) {
    int o = nt * 16 + cl;
    b2v[nt]  = b2e[o];
    w3c0[nt] = w3e[o * 2 + 0];
    w3c1[nt] = w3e[o * 2 + 1];
  }
#pragma unroll
  for (int r = 0; r < 8; ++r) {
    float v0 = 0.f, v1 = 0.f;
#pragma unroll
    for (int nt = 0; nt < 16; ++nt) {
      float h = fmaxf(acc2[nt][r] + b2v[nt], 0.f);
      v0 += h * w3c0[nt];
      v1 += h * w3c1[nt];
    }
#pragma unroll
    for (int m = 1; m < 16; m <<= 1) {   // reduce within each 16-lane half
      v0 += __shfl_xor(v0, m, 32);
      v1 += __shfl_xor(v1, m, 32);
    }
    if (cl == 0) {                       // lane 0 -> row r, lane 16 -> row r+8
      int row = row0 + wave * 16 + r + hi8;
      float gwv = gate_w[(size_t)row * NE + e];
      atomicAdd(&out_acc[row * 2 + 0], gwv * (v0 + b3[e * 2 + 0]));
      atomicAdd(&out_acc[row * 2 + 1], gwv * (v1 + b3[e * 2 + 1]));
    }
  }
}

// pred_mean = acc[:,0]; pred_var = softplus(acc[:,1]) + 1e-6
__global__ __launch_bounds__(256) void final_kernel(const float* __restrict__ acc,
                                                    float* __restrict__ out) {
  int b = blockIdx.x * 256 + threadIdx.x;
  if (b >= NB) return;
  float m = acc[b * 2 + 0];
  float v = acc[b * 2 + 1];
  float sp = fmaxf(v, 0.f) + log1pf(expf(-fabsf(v)));
  out[b]      = m;
  out[NB + b] = sp + 1e-6f;
}

// ---------------------------------------------------------------------------

extern "C" void kernel_launch(void* const* d_in, const int* in_sizes, int n_in,
                              void* d_out, int out_size, void* d_ws, size_t ws_size,
                              hipStream_t stream) {
  const float* x      = (const float*)d_in[0];
  const float* g_ln_g = (const float*)d_in[1];
  const float* g_ln_b = (const float*)d_in[2];
  const float* g_w1   = (const float*)d_in[3];
  const float* g_b1   = (const float*)d_in[4];
  const float* g_w2   = (const float*)d_in[5];
  const float* g_b2   = (const float*)d_in[6];
  const float* e_ln_g = (const float*)d_in[7];
  const float* e_ln_b = (const float*)d_in[8];
  const float* e_w1   = (const float*)d_in[9];
  const float* e_b1   = (const float*)d_in[10];
  const float* e_w2   = (const float*)d_in[11];
  const float* e_b2   = (const float*)d_in[12];
  const float* e_w3   = (const float*)d_in[13];
  const float* e_b3   = (const float*)d_in[14];
  float* out = (float*)d_out;

  // workspace layout (bytes)
  unsigned char* ws = (unsigned char*)d_ws;
  bf16*  xn_bf   = (bf16*)(ws);                              //  8 MB
  bf16*  gw1_bf  = (bf16*)(ws + 8388608);                    //  0.5 MB (256,1024)
  bf16*  w1_bf   = (bf16*)(ws + 8912896);                    // 64 MB (E,2048,1024)
  bf16*  w2_bf   = (bf16*)(ws + 76021760);                   // 16 MB (E,256,2048)
  bf16*  g_bf    = (bf16*)(ws + 92798976);                   //  2 MB
  float* logits  = (float*)(ws + 94896128);                  // 256 KB
  float* gate_w  = (float*)(ws + 95158272);                  // 256 KB
  float* out_acc = (float*)(ws + 95420416);                  // 32 KB

  // 1) weight transpose + bf16 conversion: out (N,K) <- in (K,N)
  tcvt_kernel<<<dim3(NG / 32, NIN / 32, 1),    256, 0, stream>>>(g_w1, gw1_bf, NIN, NG);
  tcvt_kernel<<<dim3(NHID / 32, NIN / 32, NE), 256, 0, stream>>>(e_w1, w1_bf, NIN, NHID);
  tcvt_kernel<<<dim3(NG / 32, NHID / 32, NE),  256, 0, stream>>>(e_w2, w2_bf, NHID, NG);

  // 2) layernorm
  ln_kernel<<<NB, 256, 0, stream>>>(x, xn_bf);

  // 3) gating
  gate_gemm1<<<NB / 128, 256, 0, stream>>>(xn_bf, g_ln_g, g_ln_b, gw1_bf, g_b1, g_bf);
  gate_logits<<<(NB * NE) / 256, 256, 0, stream>>>(g_bf, g_w2, g_b2, logits);
  gate_softmax<<<NB / 256, 256, 0, stream>>>(logits, gate_w);

  // 4) experts (fused MLP chain), accumulate gated outputs
  zero_f32<<<32, 256, 0, stream>>>(out_acc, (long)NB * 2);
  dim3 egrid(NB / 128, NE);
  expert_kernel<<<egrid, 256, 0, stream>>>(xn_bf, e_ln_g, e_ln_b, w1_bf, e_b1,
                                           w2_bf, e_b2, e_w3, e_b3, gate_w, out_acc);

  // 5) mean / softplus-var
  final_kernel<<<NB / 256, 256, 0, stream>>>(out_acc, out);
}
